// GATModel_16638703305319
// MI455X (gfx1250) — compile-verified, matrix operand order checked by reading, not run on hardware
//
#include <hip/hip_runtime.h>

typedef __attribute__((ext_vector_type(16))) _Float16 v16h;
typedef __attribute__((ext_vector_type(8)))  _Float16 v8h;
typedef __attribute__((ext_vector_type(16))) __bf16   v16bf;
typedef __attribute__((ext_vector_type(8)))  __bf16   v8bf;
typedef __attribute__((ext_vector_type(8)))  float    v8f;
typedef __attribute__((ext_vector_type(4)))  float    v4f;

#define NN 4096
#define FF 64
#define DD 64
#define HH 4
#define NEG_SLOPE 0.2f

// ---------- helpers ----------
__device__ __forceinline__ float leaky(float x) { return x > 0.f ? x : NEG_SLOPE * x; }

// round-to-nearest-even f32 -> bf16 (bit manipulation; used only in prep/epilogues)
__device__ __forceinline__ __bf16 f2bf(float f) {
    union { float f; unsigned u; } c; c.f = f;
    unsigned r = (c.u + 0x7FFFu + ((c.u >> 16) & 1u)) >> 16;
    unsigned short s = (unsigned short)r;
    __bf16 out;
    __builtin_memcpy(&out, &s, 2);
    return out;
}

// 16-bit WMMA fragment K-pattern (wave32): lanes 0-15 hold K {0..7, 16..23},
// lanes 16-31 hold K {8..15, 24..31}. With K-major (contiguous) storage this is
// two 16-byte runs per lane at offsets (hi*8) and (hi*8 + 16).

// ---------- prep: x -> bf16 ----------
__global__ void k_cvt_x(const float* __restrict__ x, __bf16* __restrict__ x_bf) {
    int i = blockIdx.x * blockDim.x + threadIdx.x;
    x_bf[i] = f2bf(x[i]);
}

// ---------- prep: W[k][n] -> WT[n][k] bf16 ----------
__global__ void k_wt(const float* __restrict__ W, __bf16* __restrict__ WT, int Kdim, int Ncols) {
    int i = blockIdx.x * blockDim.x + threadIdx.x;
    if (i >= Kdim * Ncols) return;
    int k = i / Ncols, n = i % Ncols;
    WT[n * Kdim + k] = f2bf(W[i]);
}

// build a 16-element 16-bit fragment from two contiguous 8-element runs
__device__ __forceinline__ v16bf frag_bf(const __bf16* __restrict__ p) {
    v8bf a = *(const v8bf*)p;
    v8bf b = *(const v8bf*)(p + 16);
    v16bf r;
#pragma unroll
    for (int i = 0; i < 8; ++i) { r[i] = a[i]; r[8 + i] = b[i]; }
    return r;
}
__device__ __forceinline__ v16h frag_h(const _Float16* __restrict__ p) {
    v8h a = *(const v8h*)p;
    v8h b = *(const v8h*)(p + 16);
    v16h r;
#pragma unroll
    for (int i = 0; i < 8; ++i) { r[i] = a[i]; r[8 + i] = b[i]; }
    return r;
}

// ---------- kernel 1: encoder  h = relu(x @ W_enc + b_enc) ----------
// 1 wave/block, 16 rows/block. All fragment loads are 2x b128 per lane.
__global__ void k_encode(const __bf16* __restrict__ x_bf, const __bf16* __restrict__ WT,
                         const float* __restrict__ b, float* __restrict__ h,
                         __bf16* __restrict__ h_bf) {
    int lane = threadIdx.x & 31;
    int r0 = blockIdx.x * 16;
    int m = lane & 15;
    int rb = ((lane >> 4) & 1) * 8;
    int hi = (lane >> 4) & 1;
    v8f acc[4] = {};
#pragma unroll
    for (int ks = 0; ks < 2; ++ks) {
        int k0 = ks * 32;
        v16bf af = frag_bf(x_bf + (r0 + m) * FF + k0 + rb);
#pragma unroll
        for (int nt = 0; nt < 4; ++nt) {
            v16bf bfr = frag_bf(WT + (nt * 16 + m) * FF + k0 + rb);
            acc[nt] = __builtin_amdgcn_wmma_f32_16x16x32_bf16(
                false, af, false, bfr, (short)0, acc[nt], false, false);
        }
    }
#pragma unroll
    for (int nt = 0; nt < 4; ++nt) {
        int d = nt * 16 + m;
        float bias = b[d];
#pragma unroll
        for (int v = 0; v < 8; ++v) {
            int row = v + 8 * hi;
            float val = fmaxf(acc[nt][v] + bias, 0.f);
            h[(r0 + row) * DD + d] = val;
            h_bf[(r0 + row) * DD + d] = f2bf(val);
        }
    }
}

// ---------- kernel 2: projection  hpT[h][d][s] = (h @ W)[s, h*64+d] as f16 ----------
// 4 waves/block (one per head), 16 rows/block.
__global__ void k_proj(const __bf16* __restrict__ h_bf, const __bf16* __restrict__ WT,
                       _Float16* __restrict__ hpT) {
    int tid = threadIdx.x;
    int lane = tid & 31;
    int hh = tid >> 5;
    int r0 = blockIdx.x * 16;
    int m = lane & 15;
    int rb = ((lane >> 4) & 1) * 8;
    int hi = (lane >> 4) & 1;
    v8f acc[4] = {};
#pragma unroll
    for (int ks = 0; ks < 2; ++ks) {
        int k0 = ks * 32;
        v16bf af = frag_bf(h_bf + (r0 + m) * DD + k0 + rb);
#pragma unroll
        for (int nt = 0; nt < 4; ++nt) {
            v16bf bfr = frag_bf(WT + (hh * DD + nt * 16 + m) * DD + k0 + rb);
            acc[nt] = __builtin_amdgcn_wmma_f32_16x16x32_bf16(
                false, af, false, bfr, (short)0, acc[nt], false, false);
        }
    }
    // transposed f16 store: lane holds 8 consecutive s at fixed d -> one 16B store
#pragma unroll
    for (int nt = 0; nt < 4; ++nt) {
        int d = hh * DD + nt * 16 + m;
        v8h pack;
#pragma unroll
        for (int v = 0; v < 8; ++v) pack[v] = (_Float16)acc[nt][v];
        *(v8h*)(hpT + (size_t)d * NN + r0 + hi * 8) = pack;
    }
}

// ---------- kernel 3: per-node attention scores (transposed [h][n] layout) ----------
__global__ void k_scores(const _Float16* __restrict__ hpT, const float* __restrict__ a_src,
                         const float* __restrict__ a_dst, float* __restrict__ s_srcT,
                         float* __restrict__ s_dstT) {
    int idx = blockIdx.x * blockDim.x + threadIdx.x;  // 0 .. N*H-1
    int hh = idx >> 12;
    int n = idx & (NN - 1);
    float ss = 0.f, sd = 0.f;
#pragma unroll 4
    for (int d = 0; d < DD; ++d) {
        float v = (float)hpT[((size_t)(hh * DD + d)) * NN + n];
        ss += v * a_src[hh * DD + d];
        sd += v * a_dst[hh * DD + d];
    }
    s_srcT[hh * NN + n] = ss;
    s_dstT[hh * NN + n] = sd;
}

// ---------- kernel 4: per-head max of s_src ----------
// (leaky_relu is monotonic -> softmax row max = leaky(a_t + max_s b_s); per-head global)
__global__ void k_headmax(const float* __restrict__ s_srcT, float* __restrict__ headmax) {
    __shared__ float red[256];
    int tid = threadIdx.x;
    int hh = tid >> 6;
    int j = tid & 63;
    float mx = -3.4e38f;
    for (int n = j; n < NN; n += 64) mx = fmaxf(mx, s_srcT[hh * NN + n]);
    red[tid] = mx;
    __syncthreads();
    for (int off = 32; off >= 1; off >>= 1) {
        if (j < off) red[tid] = fmaxf(red[tid], red[tid + off]);
        __syncthreads();
    }
    if (j == 0) headmax[hh] = red[tid];
}

// ---------- kernel 5: attention aggregation + head mean + bias + relu ----------
// Block: 16 targets, 8 waves = (4 heads) x (2 K-halves of 2048 sources each).
// Unnormalized f16 WMMA accumulate + per-row Z; partials combined in LDS.
__global__ void k_attn(const _Float16* __restrict__ hpT, const float* __restrict__ s_srcT,
                       const float* __restrict__ s_dstT, const float* __restrict__ headmax,
                       const float* __restrict__ bias, float* __restrict__ h_out,
                       __bf16* __restrict__ hbf_out) {
    __shared__ float part[8][16][DD];   // 32 KB
    __shared__ float zsh[8][16];
    int tid = threadIdx.x;
    int lane = tid & 31;
    int w = tid >> 5;
    int hh = w & 3;
    int half = w >> 2;
    int t0 = blockIdx.x * 16;
    int m = lane & 15;
    int hi = (lane >> 4) & 1;
    int rb = hi * 8;

    const float* bp = s_srcT + hh * NN;
    float a = s_dstT[hh * NN + t0 + m];
    float mrow = leaky(a + headmax[hh]);

    v8f acc[4] = {};
    float zl = 0.f;
    const _Float16* hpTh = hpT + (size_t)hh * DD * NN;
    int sbeg = half * (NN / 2), send = sbeg + (NN / 2);

    for (int s0 = sbeg; s0 < send; s0 += 32) {
        // A fragment: softmax weights from 4x b128 contiguous f32 loads + v_exp
        v4f b0a = *(const v4f*)(bp + s0 + rb);
        v4f b0b = *(const v4f*)(bp + s0 + rb + 4);
        v4f b1a = *(const v4f*)(bp + s0 + rb + 16);
        v4f b1b = *(const v4f*)(bp + s0 + rb + 20);
        v16h af;
#pragma unroll
        for (int i = 0; i < 4; ++i) {
            float w0 = __expf(leaky(a + b0a[i]) - mrow);
            float w1 = __expf(leaky(a + b0b[i]) - mrow);
            float w2 = __expf(leaky(a + b1a[i]) - mrow);
            float w3 = __expf(leaky(a + b1b[i]) - mrow);
            zl += (w0 + w1) + (w2 + w3);
            af[i] = (_Float16)w0;
            af[4 + i] = (_Float16)w1;
            af[8 + i] = (_Float16)w2;
            af[12 + i] = (_Float16)w3;
        }
        // B fragments: hpT rows, two contiguous 16B f16 loads per lane per tile
#pragma unroll
        for (int nt = 0; nt < 4; ++nt) {
            const _Float16* p = hpTh + (size_t)(nt * 16 + m) * NN + s0 + rb;
            v16h bfr = frag_h(p);
            __builtin_prefetch(p + 256, 0, 1);  // next chunk (L2-resident; global_prefetch)
            acc[nt] = __builtin_amdgcn_wmma_f32_16x16x32_f16(
                false, af, false, bfr, (short)0, acc[nt], false, false);
        }
    }
    // Z per target row: lanes l and l^16 hold complementary K halves of row (l&15)
    float ztot = zl + __shfl_xor(zl, 16);
    if (lane < 16) zsh[w][m] = ztot;
#pragma unroll
    for (int nt = 0; nt < 4; ++nt)
#pragma unroll
        for (int v = 0; v < 8; ++v)
            part[w][v + 8 * hi][nt * 16 + m] = acc[nt][v];
    __syncthreads();
    // combine halves, normalize per head, mean over heads + bias + relu
    for (int i = tid; i < 16 * DD; i += 256) {
        int row = i >> 6;
        int d = i & (DD - 1);
        float s = 0.f;
#pragma unroll
        for (int h2 = 0; h2 < 4; ++h2)
            s += (part[h2][row][d] + part[h2 + 4][row][d]) /
                 (zsh[h2][row] + zsh[h2 + 4][row]);
        s = fmaxf(0.25f * s + bias[d], 0.f);
        h_out[(t0 + row) * DD + d] = s;
        hbf_out[(t0 + row) * DD + d] = f2bf(s);
    }
}

// ---------- kernel 6: actor/critic heads ----------
__global__ void k_heads(const float* __restrict__ h, const float* __restrict__ W_act,
                        const float* __restrict__ b_act, const float* __restrict__ W_cri,
                        const float* __restrict__ b_cri, float* __restrict__ out) {
    int n = blockIdx.x * blockDim.x + threadIdx.x;
    if (n >= NN) return;
    float l0 = b_act[0], l1 = b_act[1], vv = b_cri[0];
#pragma unroll 4
    for (int d = 0; d < DD; ++d) {
        float hv = h[n * DD + d];
        l0 += hv * W_act[d * 2 + 0];
        l1 += hv * W_act[d * 2 + 1];
        vv += hv * W_cri[d];
    }
    l0 = fminf(fmaxf(l0, -5.f), 5.f);
    l1 = fminf(fmaxf(l1, -5.f), 5.f);
    l1 = fabsf(l1);
    out[n * 2 + 0] = l0;
    out[n * 2 + 1] = l1;
    out[2 * NN + n] = vv;
}

extern "C" void kernel_launch(void* const* d_in, const int* in_sizes, int n_in,
                              void* d_out, int out_size, void* d_ws, size_t ws_size,
                              hipStream_t stream) {
    const float* x      = (const float*)d_in[0];
    const float* W_enc  = (const float*)d_in[1];
    const float* b_enc  = (const float*)d_in[2];
    const float* W1     = (const float*)d_in[3];
    const float* a_src1 = (const float*)d_in[4];
    const float* a_dst1 = (const float*)d_in[5];
    const float* b1     = (const float*)d_in[6];
    const float* W2     = (const float*)d_in[7];
    const float* a_src2 = (const float*)d_in[8];
    const float* a_dst2 = (const float*)d_in[9];
    const float* b2     = (const float*)d_in[10];
    const float* W_act  = (const float*)d_in[11];
    const float* b_act  = (const float*)d_in[12];
    const float* W_cri  = (const float*)d_in[13];
    const float* b_cri  = (const float*)d_in[14];
    float* out = (float*)d_out;

    char* ws = (char*)d_ws;
    float*    h      = (float*)   (ws + 0);        // 1 MiB
    __bf16*   h_bf   = (__bf16*)  (ws + 1048576);  // 512 KiB
    _Float16* hpT    = (_Float16*)(ws + 1572864);  // 2 MiB
    float*    s_srcT = (float*)   (ws + 3670016);  // 64 KiB
    float*    s_dstT = (float*)   (ws + 3735552);  // 64 KiB
    float*    hmax   = (float*)   (ws + 3801088);  // 256 B
    __bf16*   x_bf   = (__bf16*)  (ws + 3801344);  // 512 KiB
    __bf16*   WencT  = (__bf16*)  (ws + 4325632);  // 8 KiB
    __bf16*   W1T    = (__bf16*)  (ws + 4333824);  // 32 KiB
    __bf16*   W2T    = (__bf16*)  (ws + 4366592);  // 32 KiB

    // prep: one-time conversions / transposes (tiny)
    k_cvt_x<<<(NN * FF) / 256, 256, 0, stream>>>(x, x_bf);
    k_wt<<<(FF * DD + 255) / 256, 256, 0, stream>>>(W_enc, WencT, FF, DD);
    k_wt<<<(DD * HH * DD + 255) / 256, 256, 0, stream>>>(W1, W1T, DD, HH * DD);
    k_wt<<<(DD * HH * DD + 255) / 256, 256, 0, stream>>>(W2, W2T, DD, HH * DD);

    // encoder
    k_encode<<<NN / 16, 32, 0, stream>>>(x_bf, WencT, b_enc, h, h_bf);

    // GAT layer 1
    k_proj<<<NN / 16, 128, 0, stream>>>(h_bf, W1T, hpT);
    k_scores<<<(NN * HH) / 256, 256, 0, stream>>>(hpT, a_src1, a_dst1, s_srcT, s_dstT);
    k_headmax<<<1, 256, 0, stream>>>(s_srcT, hmax);
    k_attn<<<NN / 16, 256, 0, stream>>>(hpT, s_srcT, s_dstT, hmax, b1, h, h_bf);

    // GAT layer 2
    k_proj<<<NN / 16, 128, 0, stream>>>(h_bf, W2T, hpT);
    k_scores<<<(NN * HH) / 256, 256, 0, stream>>>(hpT, a_src2, a_dst2, s_srcT, s_dstT);
    k_headmax<<<1, 256, 0, stream>>>(s_srcT, hmax);
    k_attn<<<NN / 16, 256, 0, stream>>>(hpT, s_srcT, s_dstT, hmax, b2, h, h_bf);

    // heads
    k_heads<<<NN / 256, 256, 0, stream>>>(h, W_act, b_act, W_cri, b_cri, out);
}
